// Linear4bit_91096256348593
// MI455X (gfx1250) — compile-verified, matrix operand order checked by reading, not run on hardware
//
#include <hip/hip_runtime.h>

// Problem dims (fixed by the reference): x[8,2048,4096] f32, W[4096,4096] int4-coded,
// scales[4096,64], bias[4096], delta_w[4096,4096]. out = x @ (deq(W)+2*delta)^T + bias.
#define M_DIM 16384
#define N_DIM 4096
#define K_DIM 4096
#define GROUPQ 64
#define SCALING 2.0f

// GEMM tiling
#define BM 128
#define BN 128
#define BK 64
#define LDSTR 72          // LDS row stride in bf16 (64 data + 8 pad; 144B keeps 16B align)
#define NBUF 4            // quad-buffered LDS, prefetch distance 2
#define NIT (K_DIM / BK)  // 64 K-steps
#define TILE_B (BM * LDSTR * 2)  // bytes per A (==B) tile buffer: 18432

typedef __attribute__((ext_vector_type(16))) __bf16 v16bf;
typedef __attribute__((ext_vector_type(8)))  float  v8f;

union Frag {
  v16bf v;
  uint4 q[2];
};

union Pack8 {
  __bf16 e[8];
  uint4 q;
};

// ---------------------------------------------------------------------------
// Pass 1: W_eff[n,k] = (code - 8) * scale[n, k/64] + 2.0 * delta_w[n,k]  -> bf16
// ---------------------------------------------------------------------------
__global__ __launch_bounds__(256) void prep_w_kernel(
    const int* __restrict__ codes, const float* __restrict__ scales,
    const float* __restrict__ delta, __bf16* __restrict__ wout)
{
  const size_t i = ((size_t)blockIdx.x * 256u + threadIdx.x) * 8u;
  const size_t row = i / K_DIM;
  const size_t k   = i % K_DIM;
  const float s = scales[row * (K_DIM / GROUPQ) + (k / GROUPQ)];

  const int4   c0 = *(const int4*)(codes + i);
  const int4   c1 = *(const int4*)(codes + i + 4);
  const float4 d0 = *(const float4*)(delta + i);
  const float4 d1 = *(const float4*)(delta + i + 4);

  Pack8 r;
  r.e[0] = (__bf16)((float)(c0.x - 8) * s + SCALING * d0.x);
  r.e[1] = (__bf16)((float)(c0.y - 8) * s + SCALING * d0.y);
  r.e[2] = (__bf16)((float)(c0.z - 8) * s + SCALING * d0.z);
  r.e[3] = (__bf16)((float)(c0.w - 8) * s + SCALING * d0.w);
  r.e[4] = (__bf16)((float)(c1.x - 8) * s + SCALING * d1.x);
  r.e[5] = (__bf16)((float)(c1.y - 8) * s + SCALING * d1.y);
  r.e[6] = (__bf16)((float)(c1.z - 8) * s + SCALING * d1.z);
  r.e[7] = (__bf16)((float)(c1.w - 8) * s + SCALING * d1.w);
  *(uint4*)(wout + i) = r.q;
}

// ---------------------------------------------------------------------------
// Pass 2: x f32 -> bf16 (x_bf16 128MB + W_bf16 32MB fit the 192MB global L2,
// so GEMM tile re-reads stay on-die).
// ---------------------------------------------------------------------------
__global__ __launch_bounds__(256) void prep_x_kernel(
    const float* __restrict__ x, __bf16* __restrict__ xout)
{
  const size_t i = ((size_t)blockIdx.x * 256u + threadIdx.x) * 8u;
  const float4 a0 = *(const float4*)(x + i);
  const float4 a1 = *(const float4*)(x + i + 4);
  Pack8 r;
  r.e[0] = (__bf16)a0.x;  r.e[1] = (__bf16)a0.y;
  r.e[2] = (__bf16)a0.z;  r.e[3] = (__bf16)a0.w;
  r.e[4] = (__bf16)a1.x;  r.e[5] = (__bf16)a1.y;
  r.e[6] = (__bf16)a1.z;  r.e[7] = (__bf16)a1.w;
  *(uint4*)(xout + i) = r.q;
}

// Issue 4 contiguous 16B async global->LDS copies (one 64B chunk of a tile row).
// GVS mode: global addr = SADDR(sgpr base) + VADDR(32b byte offset) + IOFFSET.
// The same IOFFSET is added to the LDS address, which is correct here because
// the 4 chunks are contiguous in BOTH global memory and LDS.
#define ASYNC_TILE64(ldsaddr, voff, sbase)                                   \
  asm volatile("global_load_async_to_lds_b128 %0, %1, %2\n\t"                \
               "global_load_async_to_lds_b128 %0, %1, %2 offset:16\n\t"      \
               "global_load_async_to_lds_b128 %0, %1, %2 offset:32\n\t"      \
               "global_load_async_to_lds_b128 %0, %1, %2 offset:48"          \
               :: "v"(ldsaddr), "v"(voff), "s"(sbase) : "memory")

// ---------------------------------------------------------------------------
// Pass 3: bf16 WMMA GEMM, quad-buffered async-to-LDS pipeline (distance 2,
// ONE barrier per K-step). Block = 256 threads = 8 waves (wave32).
// Tile 128x128, K-step 64. Wave (wm,wn) owns 32x64 = 2x4 WMMA 16x16 tiles.
// Fragment layout per ISA 16-bit A-matrix 16x32:
//   lane L: row = L&15, khalf = (L>>4)*8
//   VGPR0-3 <- K = khalf + 0..7 ; VGPR4-7 <- K = khalf + 16..23
// ---------------------------------------------------------------------------
__global__ __launch_bounds__(256) void gemm_bf16_wmma_kernel(
    const __bf16* __restrict__ A,   // [M_DIM, K_DIM] bf16 (x)
    const __bf16* __restrict__ B,   // [N_DIM, K_DIM] bf16 (W_eff)
    const float*  __restrict__ bias,
    float* __restrict__ out)        // [M_DIM, N_DIM] f32
{
  __shared__ __bf16 lA[NBUF][BM * LDSTR];
  __shared__ __bf16 lB[NBUF][BN * LDSTR];

  const int tid   = threadIdx.x;
  const int bm0   = blockIdx.y * BM;
  const int bn0   = blockIdx.x * BN;
  const int wave  = tid >> 5;
  const int lane  = tid & 31;
  const int wm    = wave & 3;    // 0..3 -> which 32-row slab
  const int wn    = wave >> 2;   // 0..1 -> which 64-col slab
  const int lrow  = lane & 15;
  const int lhalf = lane >> 4;   // 0/1

  // Async staging mapping: thread -> (row, one 64B chunk = 4x16B).
  const int srow = tid >> 1;         // 0..127
  const int scb  = (tid & 1) * 4;    // 16B-chunk base: 0 or 4

  unsigned voffA = (unsigned)((size_t)(bm0 + srow) * K_DIM * 2 + scb * 16);
  unsigned voffB = (unsigned)((size_t)(bn0 + srow) * K_DIM * 2 + scb * 16);
  const unsigned long long baseA = (unsigned long long)(uintptr_t)A;
  const unsigned long long baseB = (unsigned long long)(uintptr_t)B;

  const unsigned ldsA0 = (unsigned)(uintptr_t)&lA[0][srow * LDSTR + scb * 8];
  const unsigned ldsB0 = (unsigned)(uintptr_t)&lB[0][srow * LDSTR + scb * 8];

  v8f acc[2][4];
#pragma unroll
  for (int tm = 0; tm < 2; ++tm)
#pragma unroll
    for (int tn = 0; tn < 4; ++tn) {
      v8f z = {};
      acc[tm][tn] = z;
    }

  // One K-step of WMMA work out of LDS buffer `buf`.
  auto compute_tile = [&](int buf) {
#pragma unroll
    for (int kk = 0; kk < BK; kk += 32) {
      Frag af[2];
      Frag bq[4];
#pragma unroll
      for (int tm = 0; tm < 2; ++tm) {
        const __bf16* p = &lA[buf][(wm * 32 + tm * 16 + lrow) * LDSTR + kk + lhalf * 8];
        af[tm].q[0] = *(const uint4*)(p);        // K = khalf + 0..7
        af[tm].q[1] = *(const uint4*)(p + 16);   // K = khalf + 16..23
      }
#pragma unroll
      for (int tn = 0; tn < 4; ++tn) {
        const __bf16* p = &lB[buf][(wn * 64 + tn * 16 + lrow) * LDSTR + kk + lhalf * 8];
        bq[tn].q[0] = *(const uint4*)(p);
        bq[tn].q[1] = *(const uint4*)(p + 16);
      }
#pragma unroll
      for (int tm = 0; tm < 2; ++tm)
#pragma unroll
        for (int tn = 0; tn < 4; ++tn) {
          acc[tm][tn] = __builtin_amdgcn_wmma_f32_16x16x32_bf16(
              /*neg_a=*/false, af[tm].v,
              /*neg_b=*/false, bq[tn].v,
              /*c_mod=*/(short)0, acc[tm][tn],
              /*reuse_a=*/false, /*reuse_b=*/false);
        }
    }
  };

  // Prime buffers 0 and 1 (8 async b128 loads each).
  ASYNC_TILE64(ldsA0, voffA, baseA);
  ASYNC_TILE64(ldsB0, voffB, baseB);
  voffA += BK * 2;  voffB += BK * 2;
  ASYNC_TILE64(ldsA0 + 1u * TILE_B, voffA, baseA);
  ASYNC_TILE64(ldsB0 + 1u * TILE_B, voffB, baseB);
  voffA += BK * 2;  voffB += BK * 2;

  // Steady state: branch-free body, ONE barrier per K-step.
  // Iteration i: retire buf i&3 (in-order async completion => wait <= 8),
  // barrier, prefetch buf (i+2)&3 (= (i-2)&3, all reads of it finished before
  // the barrier of iteration i-1), compute buf i&3.
  for (int i = 0; i < NIT - 2; ++i) {
    asm volatile("s_wait_asynccnt 0x8" ::: "memory");
    __syncthreads();
    const unsigned pre = (unsigned)((i + 2) & (NBUF - 1)) * TILE_B;
    ASYNC_TILE64(ldsA0 + pre, voffA, baseA);
    ASYNC_TILE64(ldsB0 + pre, voffB, baseB);
    voffA += BK * 2;  voffB += BK * 2;
    compute_tile(i & (NBUF - 1));
  }
  // Peeled tail (no prefetch): wait counts become compile-time exact.
  asm volatile("s_wait_asynccnt 0x8" ::: "memory");
  __syncthreads();
  compute_tile((NIT - 2) & (NBUF - 1));
  asm volatile("s_wait_asynccnt 0x0" ::: "memory");
  __syncthreads();
  compute_tile((NIT - 1) & (NBUF - 1));

  // Epilogue: C/D layout -> M = vgpr + 8*lhalf, N = lane&15 (per 16x16 tile).
#pragma unroll
  for (int tn = 0; tn < 4; ++tn) {
    const int n = bn0 + wn * 64 + tn * 16 + lrow;
    const float bv = bias[n];
#pragma unroll
    for (int tm = 0; tm < 2; ++tm) {
      const int mbase = bm0 + wm * 32 + tm * 16 + lhalf * 8;
#pragma unroll
      for (int v = 0; v < 8; ++v) {
        out[(size_t)(mbase + v) * N_DIM + n] = acc[tm][tn][v] + bv;
      }
    }
  }
}

// ---------------------------------------------------------------------------
extern "C" void kernel_launch(void* const* d_in, const int* in_sizes, int n_in,
                              void* d_out, int out_size, void* d_ws, size_t ws_size,
                              hipStream_t stream) {
  const float* x        = (const float*)d_in[0];
  const int*   w_codes  = (const int*)d_in[1];
  const float* scales   = (const float*)d_in[2];
  const float* bias     = (const float*)d_in[3];
  const float* delta_w  = (const float*)d_in[4];
  float* out = (float*)d_out;

  __bf16* ws_w = (__bf16*)d_ws;                                                   // N*K bf16 (32MB)
  __bf16* ws_x = (__bf16*)((char*)d_ws + (size_t)N_DIM * K_DIM * sizeof(__bf16)); // M*K bf16 (128MB)

  // Pass 1: fused int4-dequant + LoRA delta -> bf16 W_eff
  prep_w_kernel<<<(N_DIM * (size_t)K_DIM) / (8 * 256), 256, 0, stream>>>(
      w_codes, scales, delta_w, ws_w);

  // Pass 2: x f32 -> bf16
  prep_x_kernel<<<((size_t)M_DIM * K_DIM) / (8 * 256), 256, 0, stream>>>(x, ws_x);

  // Pass 3: WMMA GEMM + bias
  dim3 grid(N_DIM / BN, M_DIM / BM);
  gemm_bf16_wmma_kernel<<<grid, 256, 0, stream>>>(ws_x, ws_w, bias, out);
}
